// _MRNN_46935402611163
// MI455X (gfx1250) — matrix operand order repaired and checked by reading, not compiled, not run on hardware
//
#include <hip/hip_runtime.h>

#define TT 256
#define NB 64
#define FF 64
#define HH 64
#define GD 192          // 3*H
#define NSEQ (NB*FF)    // 4096

typedef __bf16 bf16;
typedef __attribute__((ext_vector_type(16))) __bf16 v16bf;
typedef __attribute__((ext_vector_type(8)))  float  v8f;

// ---- A fragment (16x32 bf16, CDNA5 ISA §7.12.2), row-major source ----
// lane L: row = mtile*16 + (L&15); elements [0..15] = K kb+16*(L>>4) .. kb+16*(L>>4)+15
__device__ __forceinline__ v16bf load_a_frag(const bf16* __restrict__ src, int ld,
                                             int mtile, int kstep, int lane) {
  const int half = lane >> 4;
  const bf16* row = src + (mtile * 16 + (lane & 15)) * ld + kstep * 32 + half * 16;
  v16bf a;
#pragma unroll
  for (int p = 0; p < 8; ++p) {
    a[p] = row[p];
    a[8 + p] = row[8 + p];
  }
  return a;
}

// ---- B fragment from pre-swizzled LDS: [frag][lane][16 bf16] contiguous ----
__device__ __forceinline__ v16bf load_b_sw(const bf16* __restrict__ src, int frag, int lane) {
  return *(const v16bf*)(src + ((frag << 5) + lane) * 16);
}

// swizzle target index math: element i of a 64x(16*nfrag... ) fragment pool maps to
// frag = i>>9, lane = (i>>4)&31, e = i&15 ; k = (frag&1)*32 + ((lane>>4)<<4) + e ;
// n = (frag>>1)*16 + (lane&15)
__device__ __forceinline__ void frag_coords(int i, int& k, int& n) {
  const int frag = i >> 9;
  const int lidx = (i >> 4) & 31;
  const int e = i & 15;
  k = (frag & 1) * 32 + ((lidx >> 4) << 4) + e;
  n = (frag >> 1) * 16 + (lidx & 15);
}

__device__ __forceinline__ void store_acc(float* __restrict__ gh, const v8f& a,
                                          int mt, int nt, int lane) {
  const int mrow = mt * 16 + 8 * (lane >> 4);
  const int ncol = nt * 16 + (lane & 15);
#pragma unroll
  for (int r = 0; r < 8; ++r) gh[(mrow + r) * GD + ncol] = a[r];
}

__device__ __forceinline__ float sigmoidf_(float x) { return 1.f / (1.f + __expf(-x)); }
__device__ __forceinline__ float tanhf_(float x) {
  float e = __expf(2.f * x);
  return (e - 1.f) / (e + 1.f);
}

#define WMMA_BF16(A, Bv, C) \
  __builtin_amdgcn_wmma_f32_16x16x32_bf16(false, (A), false, (Bv), (short)0, (C), false, false)

// =====================================================================
// Kernel 1: GRU scan, one direction per blockIdx.y, blockIdx.x = batch b.
// w_hh kept as register-resident B fragments per wave across all T steps.
// =====================================================================
__global__ __launch_bounds__(256) void gru_scan_kernel(
    const float* __restrict__ Xf, const float* __restrict__ Mf, const float* __restrict__ Df,
    const float* __restrict__ Xb, const float* __restrict__ Mb, const float* __restrict__ Db,
    const float* __restrict__ wihf, const float* __restrict__ whhf,
    const float* __restrict__ bihf, const float* __restrict__ bhhf,
    const float* __restrict__ wihb, const float* __restrict__ whhb,
    const float* __restrict__ bihb, const float* __restrict__ bhhb,
    const float* __restrict__ Wproj, float* __restrict__ estf, float* __restrict__ estb) {
  extern __shared__ char smem[];
  float* gh = (float*)smem;                                  // 64*192 f32
  bf16* hbf = (bf16*)(smem + 64 * GD * 4);                   // 64*64 bf16
  bf16* whh = (bf16*)(smem + 64 * GD * 4 + 64 * 64 * 2);     // 24 frags * 512 bf16
  float* wih = (float*)(smem + 64 * GD * 4 + 64 * 64 * 2 + 64 * GD * 2);
  float* bias = wih + GD * 3;
  float* wp = bias + GD;
  float* xv = wp + 64;
  float* part = xv + 3 * 64;

  const int dir = blockIdx.y;
  const float* X = dir ? Xb : Xf;
  const float* Mm = dir ? Mb : Mf;
  const float* D = dir ? Db : Df;
  const float* w_ih = dir ? wihb : wihf;
  const float* w_hh = dir ? whhb : whhf;
  const float* b_ih = dir ? bihb : bihf;
  const float* b_hh = dir ? bhhb : bhhf;
  const float* wproj = Wproj + dir * HH;
  float* est = dir ? estb : estf;

  const int tid = threadIdx.x;
  const int b = blockIdx.x;
  const int lane = tid & 31;
  const int row = tid >> 2;        // gate-thread row 0..63
  const int cb = (tid & 3) * 16;   // gate-thread column base

  // Stage w_hh as swizzled B fragments: B(k,g) = w_hh[g*H + k]
  for (int i = tid; i < 64 * GD; i += 256) {
    int k, g;
    frag_coords(i, k, g);
    whh[i] = (bf16)w_hh[g * HH + k];
  }
  for (int i = tid; i < GD * 3; i += 256) wih[i] = w_ih[i];
  for (int i = tid; i < GD; i += 256) bias[i] = b_ih[i] + b_hh[i];
  if (tid < 64) wp[tid] = wproj[tid];
  for (int i = tid; i < 64 * 64 / 2; i += 256) ((unsigned*)hbf)[i] = 0u;

  float h_own[16];
#pragma unroll
  for (int i = 0; i < 16; ++i) h_own[i] = 0.f;
  __syncthreads();

  // Wave-uniform retiling: wave 2g -> (mt0..3 @ nt=3g) + (mt0..1 @ nt=3g+1)
  //                        wave 2g+1 -> (mt2..3 @ nt=3g+1) + (mt0..3 @ nt=3g+2)
  const int waveu = __builtin_amdgcn_readfirstlane(tid >> 5);
  const int grpu = waveu >> 1, subu = waveu & 1;
  const int n0 = subu ? (3 * grpu + 1) : (3 * grpu);
  const int n1 = n0 + 1;
  // register-resident, loop-invariant B fragments (32 VGPRs)
  const v16bf B00 = load_b_sw(whh, n0 * 2 + 0, lane);
  const v16bf B01 = load_b_sw(whh, n0 * 2 + 1, lane);
  const v16bf B10 = load_b_sw(whh, n1 * 2 + 0, lane);
  const v16bf B11 = load_b_sw(whh, n1 * 2 + 1, lane);

  for (int t = 0; t < TT; ++t) {
    // stage x = [X, M, D] for this step
    const int base = (b * TT + t) * FF;
    if (tid < 64) xv[tid] = X[base + tid];
    else if (tid < 128) xv[tid] = Mm[base + tid - 64];
    else if (tid < 192) xv[tid] = D[base + tid - 128];

    // gh = h(bf16) @ w_hh^T(bf16): 6 output tiles per wave, K=64 in 2 WMMA steps
    v8f acc0 = {}, acc1 = {}, acc2 = {}, acc3 = {}, acc4 = {}, acc5 = {};
#pragma unroll
    for (int ks = 0; ks < 2; ++ks) {
      const v16bf A0 = load_a_frag(hbf, HH, 0, ks, lane);
      const v16bf A1 = load_a_frag(hbf, HH, 1, ks, lane);
      const v16bf A2 = load_a_frag(hbf, HH, 2, ks, lane);
      const v16bf A3 = load_a_frag(hbf, HH, 3, ks, lane);
      const v16bf Bd0 = ks ? B01 : B00;
      const v16bf Bd1 = ks ? B11 : B10;
      if (subu == 0) {
        acc0 = WMMA_BF16(A0, Bd0, acc0);
        acc1 = WMMA_BF16(A1, Bd0, acc1);
        acc2 = WMMA_BF16(A2, Bd0, acc2);
        acc3 = WMMA_BF16(A3, Bd0, acc3);
        acc4 = WMMA_BF16(A0, Bd1, acc4);
        acc5 = WMMA_BF16(A1, Bd1, acc5);
      } else {
        acc0 = WMMA_BF16(A2, Bd0, acc0);
        acc1 = WMMA_BF16(A3, Bd0, acc1);
        acc2 = WMMA_BF16(A0, Bd1, acc2);
        acc3 = WMMA_BF16(A1, Bd1, acc3);
        acc4 = WMMA_BF16(A2, Bd1, acc4);
        acc5 = WMMA_BF16(A3, Bd1, acc5);
      }
    }
    store_acc(gh, acc0, subu ? 2 : 0, n0, lane);
    store_acc(gh, acc1, subu ? 3 : 1, n0, lane);
    store_acc(gh, acc2, subu ? 0 : 2, subu ? n1 : n0, lane);
    store_acc(gh, acc3, subu ? 1 : 3, subu ? n1 : n0, lane);
    store_acc(gh, acc4, subu ? 2 : 0, n1, lane);
    store_acc(gh, acc5, subu ? 3 : 1, n1, lane);
    __syncthreads();  // S1: gh, xv ready

    // GRU gate math (torch GRUCell), h kept in registers
    const float Xv = xv[row], Mv = xv[64 + row], Dv = xv[128 + row];
    float pest = 0.f;
#pragma unroll
    for (int i = 0; i < 16; ++i) {
      const int j = cb + i;
      const float gr = bias[j] + Xv * wih[j * 3] + Mv * wih[j * 3 + 1] + Dv * wih[j * 3 + 2] +
                       gh[row * GD + j];
      const float gz = bias[64 + j] + Xv * wih[(64 + j) * 3] + Mv * wih[(64 + j) * 3 + 1] +
                       Dv * wih[(64 + j) * 3 + 2] + gh[row * GD + 64 + j];
      const float gni = bias[128 + j] + Xv * wih[(128 + j) * 3] + Mv * wih[(128 + j) * 3 + 1] +
                        Dv * wih[(128 + j) * 3 + 2];
      const float r_ = sigmoidf_(gr);
      const float z_ = sigmoidf_(gz);
      const float nn = tanhf_(gni + r_ * gh[row * GD + 128 + j]);
      const float hnew = (1.f - z_) * nn + z_ * h_own[i];
      h_own[i] = hnew;
      pest += hnew * wp[j];
    }
    part[row * 4 + (tid & 3)] = pest;
    __syncthreads();  // S2: part ready, gh consumed

    // publish h for next step's WMMA, and emit est[t] (folded W_proj dot)
#pragma unroll
    for (int i = 0; i < 16; ++i) hbf[row * HH + cb + i] = (bf16)h_own[i];
    if (tid < 64) {
      const float e = part[tid * 4] + part[tid * 4 + 1] + part[tid * 4 + 2] + part[tid * 4 + 3];
      const int tw = dir ? (TT - 1 - t) : t;   // backward stream is time-flipped
      est[tw * NSEQ + b * FF + tid] = e;
    }
    __syncthreads();  // S3: hbf ready for next iteration
  }
}

// =====================================================================
// Kernel 2: RNN_estimation + RNN_imputed + FCN regression + imputed_data + loss partials.
// Weights staged in swizzled B-fragment order (vector ds loads).
// =====================================================================
__global__ __launch_bounds__(256) void fcn_kernel(
    const float* __restrict__ X, const float* __restrict__ Mm,
    const float* __restrict__ estf, const float* __restrict__ estb,
    const float* __restrict__ bproj, const float* __restrict__ U,
    const float* __restrict__ V1, const float* __restrict__ V2,
    const float* __restrict__ beta, const float* __restrict__ Wfin,
    const float* __restrict__ bfin, float* __restrict__ out, float* __restrict__ accum) {
  extern __shared__ char smem[];
  bf16* UB = (bf16*)smem;               // swizzled B fragments
  bf16* V1B = UB + 64 * 64;
  bf16* V2B = V1B + 64 * 64;
  bf16* WfB = V2B + 64 * 64;
  bf16* Ax = WfB + 64 * 64;             // row-major A matrices
  bf16* Ai = Ax + 64 * 64;
  bf16* AmH = Ai + 64 * 64;             // M-matrix, reused as tanh(h) for matmul 2
  float* betas = (float*)(AmH + 64 * 64);
  float* bfins = betas + 64;
  float* red = bfins + 64;

  const int tid = threadIdx.x;
  const int lane = tid & 31;
  const int wave = tid >> 5;
  const int r0 = blockIdx.x * 64;       // first global row (row = b*T + t)
  const float bp = bproj[0];

  // stage weights swizzled (U,V1 with zeroed diagonal)
  for (int i = tid; i < 64 * 64; i += 256) {
    int k, n;
    frag_coords(i, k, n);
    UB[i] = (bf16)((k == n) ? 0.f : U[n * 64 + k]);
    V1B[i] = (bf16)((k == n) ? 0.f : V1[n * 64 + k]);
    V2B[i] = (bf16)V2[n * 64 + k];
    WfB[i] = (bf16)Wfin[n * 64 + k];
  }
  if (tid < 64) { betas[tid] = beta[tid]; bfins[tid] = bfin[tid]; }

  // stage A matrices: X, RNN_imputed, M (bf16, row-major)
  for (int i = tid; i < 64 * 64; i += 256) {
    const int rl = i >> 6, f = i & 63;
    const int rg = r0 + rl;
    const int idx = rg * FF + f;
    const int bb = rg >> 8, t = rg & 255;
    const float xvv = X[idx], mv = Mm[idx];
    const float rnn = estf[t * NSEQ + bb * FF + f] + estb[t * NSEQ + bb * FF + f] + bp;
    Ax[i] = (bf16)xvv;
    Ai[i] = (bf16)(mv * xvv + (1.f - mv) * rnn);
    AmH[i] = (bf16)mv;
  }
  __syncthreads();

  // matmul 1: pre = X@UB + imp@V1B + M@V2B  (16 tiles, 2 per wave, 6 WMMA each)
  v8f acc[2];
#pragma unroll
  for (int tt = 0; tt < 2; ++tt) {
    const int tile = wave * 2 + tt;
    const int mt = tile & 3, nt = tile >> 2;
    v8f a = {};
#pragma unroll
    for (int ks = 0; ks < 2; ++ks) {
      a = WMMA_BF16(load_a_frag(Ax, 64, mt, ks, lane), load_b_sw(UB, nt * 2 + ks, lane), a);
      a = WMMA_BF16(load_a_frag(Ai, 64, mt, ks, lane), load_b_sw(V1B, nt * 2 + ks, lane), a);
      a = WMMA_BF16(load_a_frag(AmH, 64, mt, ks, lane), load_b_sw(V2B, nt * 2 + ks, lane), a);
    }
    acc[tt] = a;
  }
  __syncthreads();  // AmH reads done

  // h = tanh(pre + beta) -> bf16, written into AmH region (reused)
#pragma unroll
  for (int tt = 0; tt < 2; ++tt) {
    const int tile = wave * 2 + tt;
    const int mt = tile & 3, nt = tile >> 2;
    const int mrow = mt * 16 + 8 * (lane >> 4);
    const int ncol = nt * 16 + (lane & 15);
#pragma unroll
    for (int r = 0; r < 8; ++r)
      AmH[(mrow + r) * 64 + ncol] = (bf16)tanhf_(acc[tt][r] + betas[ncol]);
  }
  __syncthreads();

  // matmul 2: fcn_pre = h @ WfB ; then elementwise epilogue + loss partials
  float sef = 0.f, ser = 0.f, msum = 0.f;
#pragma unroll
  for (int tt = 0; tt < 2; ++tt) {
    const int tile = wave * 2 + tt;
    const int mt = tile & 3, nt = tile >> 2;
    v8f a = {};
#pragma unroll
    for (int ks = 0; ks < 2; ++ks)
      a = WMMA_BF16(load_a_frag(AmH, 64, mt, ks, lane), load_b_sw(WfB, nt * 2 + ks, lane), a);
    const int mrow = mt * 16 + 8 * (lane >> 4);
    const int f = nt * 16 + (lane & 15);
#pragma unroll
    for (int r = 0; r < 8; ++r) {
      const int rg = r0 + mrow + r;
      const int idx = rg * FF + f;
      const int bb = rg >> 8, t = rg & 255;
      const float xvv = X[idx], mv = Mm[idx];
      const float fcn = a[r] + bfins[f];
      out[idx] = mv * xvv + (1.f - mv) * fcn;
      const float d1 = (fcn - xvv) * mv;
      sef += d1 * d1;
      const float rnn = estf[t * NSEQ + bb * FF + f] + estb[t * NSEQ + bb * FF + f] + bp;
      const float d2 = (rnn - xvv) * mv;
      ser += d2 * d2;
      msum += mv;
    }
  }

  // block reduction of (sef, ser, msum) then global atomic accumulate
  float vals[3] = {sef, ser, msum};
#pragma unroll
  for (int c = 0; c < 3; ++c) {
    __syncthreads();
    red[tid] = vals[c];
    __syncthreads();
    for (int s = 128; s > 0; s >>= 1) {
      if (tid < s) red[tid] += red[tid + s];
      __syncthreads();
    }
    if (tid == 0) atomicAdd(&accum[c], red[0]);
  }
}

__global__ void init_accum_kernel(float* accum) {
  if (threadIdx.x < 3) accum[threadIdx.x] = 0.f;
}

__global__ void finalize_kernel(const float* __restrict__ accum, float* __restrict__ loss) {
  const float denom = accum[2] + 1e-12f;
  loss[0] = sqrtf(accum[0] / denom) + sqrtf(accum[1] / denom);
}

extern "C" void kernel_launch(void* const* d_in, const int* in_sizes, int n_in,
                              void* d_out, int out_size, void* d_ws, size_t ws_size,
                              hipStream_t stream) {
  const float* X_f = (const float*)d_in[0];
  const float* M_f = (const float*)d_in[1];
  const float* D_f = (const float*)d_in[2];
  const float* X_b = (const float*)d_in[3];
  const float* M_b = (const float*)d_in[4];
  const float* D_b = (const float*)d_in[5];
  const float* w_ih_f = (const float*)d_in[6];
  const float* w_hh_f = (const float*)d_in[7];
  const float* b_ih_f = (const float*)d_in[8];
  const float* b_hh_f = (const float*)d_in[9];
  const float* w_ih_b = (const float*)d_in[10];
  const float* w_hh_b = (const float*)d_in[11];
  const float* b_ih_b = (const float*)d_in[12];
  const float* b_hh_b = (const float*)d_in[13];
  const float* W_proj = (const float*)d_in[14];
  const float* b_proj = (const float*)d_in[15];
  const float* U = (const float*)d_in[16];
  const float* V1 = (const float*)d_in[17];
  const float* V2 = (const float*)d_in[18];
  const float* beta = (const float*)d_in[19];
  const float* W_fin = (const float*)d_in[20];
  const float* b_fin = (const float*)d_in[21];

  float* est_f = (float*)d_ws;                 // [T][4096] = 4 MB
  float* est_b = est_f + TT * NSEQ;            // 4 MB
  float* accum = est_b + TT * NSEQ;            // 3 floats
  float* out = (float*)d_out;

  const size_t sh1 = 64 * GD * 4 /*gh*/ + 64 * 64 * 2 /*hbf*/ + 64 * GD * 2 /*whh*/ +
                     (GD * 3 + GD + 64 + 3 * 64 + 64 * 4) * 4;
  const size_t sh2 = 7 * 64 * 64 * 2 + (64 + 64 + 256) * 4;

  gru_scan_kernel<<<dim3(NB, 2), 256, sh1, stream>>>(
      X_f, M_f, D_f, X_b, M_b, D_b, w_ih_f, w_hh_f, b_ih_f, b_hh_f,
      w_ih_b, w_hh_b, b_ih_b, b_hh_b, W_proj, est_f, est_b);
  init_accum_kernel<<<1, 64, 0, stream>>>(accum);
  fcn_kernel<<<dim3(NB * TT / 64), 256, sh2, stream>>>(
      X_f, M_f, est_f, est_b, b_proj, U, V1, V2, beta, W_fin, b_fin, out, accum);
  finalize_kernel<<<1, 1, 0, stream>>>(accum, out + NB * TT * FF);
}
